// WIKGMIL_78855599554711
// MI455X (gfx1250) — compile-verified, matrix operand order checked by reading, not run on hardware
//
#include <hip/hip_runtime.h>
#include <math.h>

// ---------------------------------------------------------------------------
// Problem constants (match reference)
// ---------------------------------------------------------------------------
#define MN   8192      // nodes
#define ED   512       // embedding dim E
#define IND  384       // input dim
#define KTOP 6

typedef __attribute__((ext_vector_type(16))) __bf16 v16bf;
typedef __attribute__((ext_vector_type(8)))  float  v8f;
typedef __attribute__((ext_vector_type(4)))  unsigned int u32x4;
typedef __attribute__((ext_vector_type(8)))  unsigned int u32x8;

union FragBF { v16bf v; unsigned int u[8]; };
union U4 { u32x4 v; unsigned int a[4]; };
union U8 { u32x8 v; unsigned int a[8]; };

__device__ __forceinline__ unsigned short f2bf(float f) {
  unsigned int u = __float_as_uint(f);
  u += 0x7FFFu + ((u >> 16) & 1u);           // round-to-nearest-even
  return (unsigned short)(u >> 16);
}
__device__ __forceinline__ unsigned int packbf(float lo, float hi) {
  return (unsigned int)f2bf(lo) | ((unsigned int)f2bf(hi) << 16);
}
__device__ __forceinline__ float lrelu(float x) { return x > 0.f ? x : 0.01f * x; }
// ISA 16-bit A/B fragment K mapping: VGPR v, lane-half hi
__device__ __forceinline__ int kmap(int v, int hi) {
  return (v < 4 ? 2 * v : 16 + 2 * (v - 4)) + hi * 8;
}

// ---------------------------------------------------------------------------
// TDM: async DMA one 16x512 bf16 tile (row stride 512 elems in memory) into
// LDS with 4-DWORD padding after each 256-DWORD row (-> 260-dword LDS stride,
// bank-conflict-free fragment reads). Wave-level; tracked by TENSORcnt.
// ---------------------------------------------------------------------------
__device__ __forceinline__ void tdm_load_tile(const unsigned* gsrc, unsigned lds_byte) {
  unsigned long long ga = (unsigned long long)(uintptr_t)gsrc;
  U4 g0;
  g0.a[0] = 1u;                                          // count=1, user mode
  g0.a[1] = lds_byte;                                    // lds_addr
  g0.a[2] = (unsigned)(ga & 0xFFFFFFFFu);                // global_addr[31:0]
  g0.a[3] = (unsigned)((ga >> 32) & 0x01FFFFFFu) | 0x80000000u; // addr[56:32] | type=2
  U8 g1;
  g1.a[0] = 0x07D10000u;  // data_size=1(2B) | pad_enable | pad_interval=7 | pad_amount=3
  g1.a[1] = 0x02000000u;  // tensor_dim0 = 512  (bits[31:16] = low16)
  g1.a[2] = 0x00100000u;  // tensor_dim1 = 16   (bits[31:16] = low16)
  g1.a[3] = 0x02000000u;  // tile_dim0   = 512  (bits[31:16])
  g1.a[4] = 0x00000010u;  // tile_dim1 = 16, tile_dim2 = 0
  g1.a[5] = 512u;         // tensor_dim0_stride[31:0]
  g1.a[6] = 0u;
  g1.a[7] = 0u;
  asm volatile("tensor_load_to_lds %0, %1" :: "s"(g0.v), "s"(g1.v) : "memory");
}

// ---------------------------------------------------------------------------
// fp32 -> packed bf16 pairs (optionally scaled)
// ---------------------------------------------------------------------------
__global__ void __launch_bounds__(256)
cvt_pack(const float* __restrict__ s, unsigned* __restrict__ d, float scale) {
  size_t i = (size_t)blockIdx.x * 256 + threadIdx.x;
  d[i] = packbf(s[2 * i] * scale, s[2 * i + 1] * scale);
}

// ---------------------------------------------------------------------------
// Generic bf16 WMMA GEMM:  C[M,N] = act(A[M,K] @ W[N,K]^T + bias)
// block = 256 thr (8 waves), tile 32x64, waves 2x4 of 16x16.
// MODE bit0 = leaky-relu, bit1 = accumulate into C   (compile-time)
// ---------------------------------------------------------------------------
template <int MODE>
__global__ void __launch_bounds__(256)
gemm_bf16(const float* __restrict__ A, const float* __restrict__ W,
          const float* __restrict__ bias, float* __restrict__ C,
          int M, int N, int K) {
  __shared__ unsigned short As[32 * 36];
  __shared__ unsigned short Bs[64 * 36];
  const int t = threadIdx.x;
  const int wave = t >> 5, lane = t & 31;
  const int r = lane & 15, hi = lane >> 4;
  const int wrow = (wave >> 2) * 16;   // 0 / 16
  const int wcol = (wave & 3) * 16;    // 0..48
  const int rowBase = blockIdx.y * 32;
  const int colBase = blockIdx.x * 64;

  v8f acc = {};
  for (int k0 = 0; k0 < K; k0 += 32) {
    __syncthreads();
    { // A tile 32x32, 4 elems/thread, coalesced fp32 -> bf16 LDS
      int rr = t >> 3, kk = (t & 7) * 4;
      const float* ap = A + (size_t)(rowBase + rr) * K + k0 + kk;
      if (k0 + 32 < K) __builtin_prefetch(ap + 32, 0, 1);
      unsigned short* dst = As + rr * 36 + kk;
#pragma unroll
      for (int i = 0; i < 4; ++i) dst[i] = f2bf(ap[i]);
    }
    { // B tile 64x32 (W row-major [N,K]), 8 elems/thread
      int nn = t >> 2, kk = (t & 3) * 8;
      const float* bp = W + (size_t)(colBase + nn) * K + k0 + kk;
      if (k0 + 32 < K) __builtin_prefetch(bp + 32, 0, 1);
      unsigned short* dst = Bs + nn * 36 + kk;
#pragma unroll
      for (int i = 0; i < 8; ++i) dst[i] = f2bf(bp[i]);
    }
    __syncthreads();
    FragBF a, b;
#pragma unroll
    for (int v = 0; v < 8; ++v) {
      int kl = kmap(v, hi);
      a.u[v] = *(const unsigned int*)&As[(wrow + r) * 36 + kl];
      b.u[v] = *(const unsigned int*)&Bs[(wcol + r) * 36 + kl];
    }
    acc = __builtin_amdgcn_wmma_f32_16x16x32_bf16(false, a.v, false, b.v,
                                                  (short)0, acc, false, false);
  }
  const int col = colBase + wcol + r;
  const float bv = bias[col];
#pragma unroll
  for (int i = 0; i < 8; ++i) {
    int row = rowBase + wrow + i + hi * 8;
    float val = acc[i] + bv;
    if (MODE & 1) val = lrelu(val);
    float* cp = C + (size_t)row * N + col;
    if (MODE & 2) *cp += val; else *cp = val;
  }
}

// ---------------------------------------------------------------------------
// Fused attention-logits + streaming top-6 over bf16 inputs.
// Block = 128 rows (8 waves x 16 rows, strict wave ownership -> no merge).
// A panel (16x512 per wave, pre-scaled bf16) preloaded into 16 fragments.
// B panel: TDM double-buffered async DMA into padded LDS; only ds_load +
// v_wmma + top-6 update remain in the 512-iteration hot loop.
// ---------------------------------------------------------------------------
__global__ void __launch_bounds__(256)
attn_topk(const unsigned* __restrict__ ehb,   // e_h * E^-0.5, packed bf16 pairs
          const unsigned* __restrict__ etb,   // e_t, packed bf16 pairs
          float* __restrict__ tkw, int* __restrict__ tki) {
  __shared__ unsigned Bs2[2][16 * 260];       // 260-dword row stride (TDM pad)
  __shared__ float tilebuf[8][16][16];
  __shared__ float tv[128][KTOP];
  __shared__ int   ti[128][KTOP];
  const int t = threadIdx.x, wave = t >> 5, lane = t & 31;
  const int r = lane & 15, hi = lane >> 4;
  const int rowBase = blockIdx.x * 128;
  const int myrow = rowBase + wave * 16 + r;
  const int NT = MN / 16;

  if (lane < 16) {
#pragma unroll
    for (int k = 0; k < KTOP; ++k) { tv[wave * 16 + lane][k] = -3.4e38f; ti[wave * 16 + lane][k] = 0; }
  }

  // Kick off tile 0 DMA (wave 0), overlap with A-panel preload below.
  if (wave == 0) {
    tdm_load_tile(etb, (unsigned)(uintptr_t)&Bs2[0][0]);
  }

  // Preload the wave's full 16x512 A panel: 16 bf16 fragments in VGPRs.
  FragBF afr[16];
#pragma unroll
  for (int kf = 0; kf < 16; ++kf) {
#pragma unroll
    for (int v = 0; v < 8; ++v) {
      afr[kf].u[v] = ehb[(size_t)myrow * (ED / 2) + kf * 16 + (kmap(v, hi) >> 1)];
    }
  }

  for (int jt = 0; jt < NT; ++jt) {
    const int buf = jt & 1;
    __syncthreads();                     // everyone done reading buf^1 (iter jt-1)
    if (wave == 0) {
      if (jt + 1 < NT)
        tdm_load_tile(etb + (size_t)(jt + 1) * 16 * (ED / 2),
                      (unsigned)(uintptr_t)&Bs2[buf ^ 1][0]);
      if (jt + 1 < NT) __builtin_amdgcn_s_wait_tensorcnt(1);  // tile jt landed
      else             __builtin_amdgcn_s_wait_tensorcnt(0);
    }
    __syncthreads();                     // publish tile jt to all waves

    v8f acc = {};
#pragma unroll
    for (int kf = 0; kf < 16; ++kf) {
      FragBF b;
#pragma unroll
      for (int v = 0; v < 8; ++v) {
        b.u[v] = Bs2[buf][r * 260 + kf * 16 + (kmap(v, hi) >> 1)];
      }
      acc = __builtin_amdgcn_wmma_f32_16x16x32_bf16(false, afr[kf].v, false, b.v,
                                                    (short)0, acc, false, false);
    }
    // wave-private staging + top-6 update (lane l owns row wave*16+l)
#pragma unroll
    for (int i = 0; i < 8; ++i) tilebuf[wave][i + hi * 8][r] = acc[i];
    if (lane < 16) {
      int lr = wave * 16 + lane;
#pragma unroll
      for (int c = 0; c < 16; ++c) {
        float val = tilebuf[wave][lane][c];
        int ms = 0; float mv = tv[lr][0];
#pragma unroll
        for (int k = 1; k < KTOP; ++k) { float x = tv[lr][k]; if (x < mv) { mv = x; ms = k; } }
        if (val > mv) { tv[lr][ms] = val; ti[lr][ms] = jt * 16 + c; }
      }
    }
  }
  __syncthreads();
  if (t < 128) {
    int grow = rowBase + t;
#pragma unroll
    for (int k = 0; k < KTOP; ++k) {
      tkw[grow * KTOP + k] = tv[t][k];
      tki[grow * KTOP + k] = ti[t][k];
    }
  }
}

// ---------------------------------------------------------------------------
// Per-row aggregation: softmax(top6) -> gather e_t rows -> tanh gate ->
// ka softmax -> e_Nh; emit e_h+e_Nh and e_h*e_Nh. One wave per row.
// ---------------------------------------------------------------------------
__global__ void __launch_bounds__(256)
aggregate(const float* __restrict__ e_h, const float* __restrict__ e_t,
          const float* __restrict__ tkw, const int* __restrict__ tki,
          float* __restrict__ es, float* __restrict__ em) {
  const int wave = threadIdx.x >> 5, lane = threadIdx.x & 31;
  const int row = blockIdx.x * 8 + wave;
  float w[KTOP]; int id[KTOP];
#pragma unroll
  for (int k = 0; k < KTOP; ++k) { w[k] = tkw[row * KTOP + k]; id[k] = tki[row * KTOP + k]; }
  float mx = w[0];
#pragma unroll
  for (int k = 1; k < KTOP; ++k) mx = fmaxf(mx, w[k]);
  float p[KTOP], s = 0.f;
#pragma unroll
  for (int k = 0; k < KTOP; ++k) { p[k] = __expf(w[k] - mx); s += p[k]; }
  float inv = 1.f / s;
#pragma unroll
  for (int k = 0; k < KTOP; ++k) p[k] *= inv;

  float eh[16];
#pragma unroll
  for (int j = 0; j < 16; ++j) eh[j] = e_h[(size_t)row * ED + lane + 32 * j];

  float ka[KTOP];
#pragma unroll
  for (int k = 0; k < KTOP; ++k) {
    float nbs = 0.f, gs = 0.f;
    const float* nr = e_t + (size_t)id[k] * ED;
#pragma unroll
    for (int j = 0; j < 16; ++j) {
      float nb = nr[lane + 32 * j];
      float ehr = p[k] * nb + (1.f - p[k]) * eh[j];
      nbs += nb;
      gs += tanhf(eh[j] + ehr);
    }
    for (int off = 16; off; off >>= 1) {
      nbs += __shfl_xor(nbs, off, 32);
      gs  += __shfl_xor(gs,  off, 32);
    }
    ka[k] = nbs * gs;
  }
  float km = ka[0];
#pragma unroll
  for (int k = 1; k < KTOP; ++k) km = fmaxf(km, ka[k]);
  float q[KTOP], qs = 0.f;
#pragma unroll
  for (int k = 0; k < KTOP; ++k) { q[k] = __expf(ka[k] - km); qs += q[k]; }
  float qi = 1.f / qs;
#pragma unroll
  for (int k = 0; k < KTOP; ++k) q[k] *= qi;

#pragma unroll
  for (int j = 0; j < 16; ++j) {
    float enh = 0.f;
#pragma unroll
    for (int k = 0; k < KTOP; ++k) enh += q[k] * e_t[(size_t)id[k] * ED + lane + 32 * j];
    size_t o = (size_t)row * ED + lane + 32 * j;
    es[o] = eh[j] + enh;
    em[o] = eh[j] * enh;
  }
}

// ---------------------------------------------------------------------------
// Small epilogue kernels
// ---------------------------------------------------------------------------
__global__ void colsum_k(const float* __restrict__ h, float* __restrict__ cs) {
  int t = threadIdx.x, rb = blockIdx.x * 128;
  float a0 = 0.f, a1 = 0.f;
  for (int rr = 0; rr < 128; ++rr) {
    const float* er = h + (size_t)(rb + rr) * ED;
    a0 += er[t]; a1 += er[t + 256];
  }
  atomicAdd(&cs[t], a0); atomicAdd(&cs[t + 256], a1);
}

__global__ void fix_h(float* __restrict__ h, const float* __restrict__ cs) {
  int i = blockIdx.x * 256 + threadIdx.x;
  int c = i & (ED - 1);
  h[i] = (h[i] + cs[c] * (1.f / MN)) * 0.5f;
}

__global__ void a2_dot(const float* __restrict__ ga, const float* __restrict__ a2w,
                       const float* __restrict__ a2b, float* __restrict__ g) {
  const int wave = threadIdx.x >> 5, lane = threadIdx.x & 31;
  const int row = blockIdx.x * 8 + wave;
  float acc = 0.f;
#pragma unroll
  for (int j = 0; j < 8; ++j) {
    int e = lane + 32 * j;
    acc += ga[(size_t)row * (ED / 2) + e] * a2w[e];
  }
  for (int off = 16; off; off >>= 1) acc += __shfl_xor(acc, off, 32);
  if (lane == 0) g[row] = acc + a2b[0];
}

__global__ void softmax_nodes(const float* __restrict__ g, float* __restrict__ att) {
  __shared__ float red[256];
  int t = threadIdx.x;
  float mx = -3.4e38f;
  for (int i = t; i < MN; i += 256) mx = fmaxf(mx, g[i]);
  red[t] = mx; __syncthreads();
  for (int s = 128; s; s >>= 1) { if (t < s) red[t] = fmaxf(red[t], red[t + s]); __syncthreads(); }
  mx = red[0]; __syncthreads();
  float sm = 0.f;
  for (int i = t; i < MN; i += 256) sm += __expf(g[i] - mx);
  red[t] = sm; __syncthreads();
  for (int s = 128; s; s >>= 1) { if (t < s) red[t] += red[t + s]; __syncthreads(); }
  float inv = 1.f / red[0];
  for (int i = t; i < MN; i += 256) att[i] = __expf(g[i] - mx) * inv;
}

__global__ void pool_k(const float* __restrict__ att, const float* __restrict__ emb,
                       float* __restrict__ pooled) {
  int t = threadIdx.x, rb = blockIdx.x * 128;
  float a0 = 0.f, a1 = 0.f;
  for (int rr = 0; rr < 128; ++rr) {
    float a = att[rb + rr];
    const float* er = emb + (size_t)(rb + rr) * ED;
    a0 += a * er[t]; a1 += a * er[t + 256];
  }
  atomicAdd(&pooled[t], a0); atomicAdd(&pooled[t + 256], a1);
}

__global__ void layernorm_k(const float* __restrict__ p, const float* __restrict__ gam,
                            const float* __restrict__ bet, float* __restrict__ out) {
  __shared__ float red[256];
  int t = threadIdx.x;
  float v0 = p[t], v1 = p[t + 256];
  red[t] = v0 + v1; __syncthreads();
  for (int s = 128; s; s >>= 1) { if (t < s) red[t] += red[t + s]; __syncthreads(); }
  float mean = red[0] * (1.f / ED); __syncthreads();
  float d0 = v0 - mean, d1 = v1 - mean;
  red[t] = d0 * d0 + d1 * d1; __syncthreads();
  for (int s = 128; s; s >>= 1) { if (t < s) red[t] += red[t + s]; __syncthreads(); }
  float inv = rsqrtf(red[0] * (1.f / ED) + 1e-5f);
  out[t]       = d0 * inv * gam[t]       + bet[t];
  out[t + 256] = d1 * inv * gam[t + 256] + bet[t + 256];
}

// ---------------------------------------------------------------------------
// Host-side orchestration
// ---------------------------------------------------------------------------
extern "C" void kernel_launch(void* const* d_in, const int* in_sizes, int n_in,
                              void* d_out, int out_size, void* d_ws, size_t ws_size,
                              hipStream_t stream) {
  (void)in_sizes; (void)n_in; (void)out_size; (void)ws_size;
  const float* x     = (const float*)d_in[0];
  const float* fc1_w = (const float*)d_in[1];
  const float* fc1_b = (const float*)d_in[2];
  const float* wh_w  = (const float*)d_in[3];
  const float* wh_b  = (const float*)d_in[4];
  const float* wt_w  = (const float*)d_in[5];
  const float* wt_b  = (const float*)d_in[6];
  const float* l1_w  = (const float*)d_in[7];
  const float* l1_b  = (const float*)d_in[8];
  const float* l2_w  = (const float*)d_in[9];
  const float* l2_b  = (const float*)d_in[10];
  const float* a1_w  = (const float*)d_in[11];
  const float* a1_b  = (const float*)d_in[12];
  const float* a2_w  = (const float*)d_in[13];
  const float* a2_b  = (const float*)d_in[14];
  const float* ln_g  = (const float*)d_in[15];
  const float* ln_b  = (const float*)d_in[16];

  float* ws = (float*)d_ws;
  size_t off = 0;
  auto carve = [&](size_t n) { float* p = ws + off; off += n; return p; };
  float*    h      = carve((size_t)MN * ED);
  float*    e_h    = carve((size_t)MN * ED);
  float*    e_t    = carve((size_t)MN * ED);
  float*    es     = carve((size_t)MN * ED);
  float*    em     = carve((size_t)MN * ED);
  float*    emb    = carve((size_t)MN * ED);
  unsigned* ehb    = (unsigned*)carve((size_t)MN * ED / 2);  // packed bf16 pairs
  unsigned* etb    = (unsigned*)carve((size_t)MN * ED / 2);
  float*    ga     = carve((size_t)MN * (ED / 2));
  float*    tkw    = carve((size_t)MN * KTOP);
  int*      tki    = (int*)carve((size_t)MN * KTOP);
  float*    gg     = carve(MN);
  float*    att    = carve(MN);
  float*    cs     = carve(ED);
  float*    pooled = carve(ED);

  dim3 gE(ED / 64, MN / 32);     // N=512 GEMMs
  dim3 gH(ED / 128, MN / 32);    // N=256 GEMM (a1)
  const int cvtBlocks = (MN * ED / 2) / 256;

  // h = lrelu(x @ fc1^T + b)
  gemm_bf16<1><<<gE, 256, 0, stream>>>(x, fc1_w, fc1_b, h, MN, ED, IND);
  // h = (h + mean_nodes(h)) / 2
  hipMemsetAsync(cs, 0, ED * sizeof(float), stream);
  colsum_k<<<MN / 128, 256, 0, stream>>>(h, cs);
  fix_h<<<(size_t)MN * ED / 256, 256, 0, stream>>>(h, cs);
  // e_h, e_t
  gemm_bf16<0><<<gE, 256, 0, stream>>>(h, wh_w, wh_b, e_h, MN, ED, ED);
  gemm_bf16<0><<<gE, 256, 0, stream>>>(h, wt_w, wt_b, e_t, MN, ED, ED);
  // bf16 copies for the attention GEMM (e_h pre-scaled by E^-0.5)
  cvt_pack<<<cvtBlocks, 256, 0, stream>>>(e_h, ehb, 0.044194173824159216f);
  cvt_pack<<<cvtBlocks, 256, 0, stream>>>(e_t, etb, 1.0f);
  // fused 8192x8192 logits + streaming top-6 (never materialized), TDM-fed
  attn_topk<<<MN / 128, 256, 0, stream>>>(ehb, etb, tkw, tki);
  // per-row gather / gate / e_Nh
  aggregate<<<MN / 8, 256, 0, stream>>>(e_h, e_t, tkw, tki, es, em);
  // emb = lrelu(es@l1^T+b) + lrelu(em@l2^T+b)
  gemm_bf16<1><<<gE, 256, 0, stream>>>(es, l1_w, l1_b, emb, MN, ED, ED);
  gemm_bf16<3><<<gE, 256, 0, stream>>>(em, l2_w, l2_b, emb, MN, ED, ED);
  // gate_nn
  gemm_bf16<1><<<gH, 256, 0, stream>>>(emb, a1_w, a1_b, ga, MN, ED / 2, ED);
  a2_dot<<<MN / 8, 256, 0, stream>>>(ga, a2_w, a2_b, gg);
  softmax_nodes<<<1, 256, 0, stream>>>(gg, att);
  // pooled + LayerNorm
  hipMemsetAsync(pooled, 0, ED * sizeof(float), stream);
  pool_k<<<MN / 128, 256, 0, stream>>>(att, emb, pooled);
  layernorm_k<<<1, 256, 0, stream>>>(pooled, ln_g, ln_b, (float*)d_out);
}
